// DConvBatchNorm_90580860272664
// MI455X (gfx1250) — compile-verified, matrix operand order checked by reading, not run on hardware
//
#include <hip/hip_runtime.h>
#include <hip/hip_bf16.h>

typedef __attribute__((ext_vector_type(16))) _Float16 v16h;
typedef __attribute__((ext_vector_type(8)))  _Float16 v8h;
typedef __attribute__((ext_vector_type(8)))  float    v8f;

#define BATCH 8
#define CIN   64
#define COUT  64
#define HH    128
#define WW    128
#define KTAP  9
#define KDIM  (CIN * KTAP)   // 576
#define TILE_M 16
#define NPIX  (BATCH * HH * WW)      // 131072
#define NTILE (NPIX / TILE_M)        // 8192

// ---------------------------------------------------------------------------
// Kernel 0: prep — zero BN accumulators, convert deform weights to f16.
// w is (COUT, CIN, 3, 3) contiguous fp32 -> wh[o][c*9+tap] f16 (same flat order)
// ---------------------------------------------------------------------------
__global__ __launch_bounds__(256) void dcn_prep_kernel(
    const float* __restrict__ w, _Float16* __restrict__ wh,
    float* __restrict__ sums) {
  int i = blockIdx.x * blockDim.x + threadIdx.x;
  if (i < 2 * COUT) sums[i] = 0.0f;
  if (i < COUT * KDIM) wh[i] = (_Float16)w[i];
}

// ---------------------------------------------------------------------------
// Kernel 1: offset (18ch) + mask (9ch) 3x3 conv, fused 2*sigmoid for mask.
// One thread per output pixel; weight addresses are wave-uniform -> SMEM loads.
// ---------------------------------------------------------------------------
__global__ __launch_bounds__(256) void dcn_offmask_kernel(
    const float* __restrict__ x,
    const float* __restrict__ off_w, const float* __restrict__ off_b,
    const float* __restrict__ mod_w, const float* __restrict__ mod_b,
    float* __restrict__ off_out, float* __restrict__ mask_out) {
  int pid = blockIdx.x * blockDim.x + threadIdx.x;
  if (pid >= NPIX) return;
  int wpix = pid % WW;
  int h    = (pid / WW) % HH;
  int b    = pid / (HH * WW);

  float acc[27];
#pragma unroll
  for (int i = 0; i < 27; ++i) acc[i] = 0.0f;

  for (int c = 0; c < CIN; ++c) {
    const float* xc = x + ((size_t)(b * CIN + c) * HH) * WW;
    float xv[9];
#pragma unroll
    for (int tap = 0; tap < 9; ++tap) {
      int yy = h + tap / 3 - 1;
      int xx = wpix + tap % 3 - 1;
      bool ok = (yy >= 0) & (yy < HH) & (xx >= 0) & (xx < WW);
      xv[tap] = ok ? xc[yy * WW + xx] : 0.0f;
    }
#pragma unroll
    for (int tap = 0; tap < 9; ++tap) {
      float v = xv[tap];
      const float* ow = off_w + (c * 9 + tap);
#pragma unroll
      for (int oc = 0; oc < 18; ++oc) acc[oc] += v * ow[oc * KDIM];
      const float* mw = mod_w + (c * 9 + tap);
#pragma unroll
      for (int oc = 0; oc < 9; ++oc) acc[18 + oc] += v * mw[oc * KDIM];
    }
  }
  for (int oc = 0; oc < 18; ++oc)
    off_out[((size_t)(b * 18 + oc) * HH + h) * WW + wpix] = acc[oc] + off_b[oc];
  for (int oc = 0; oc < 9; ++oc) {
    float z = acc[18 + oc] + mod_b[oc];
    mask_out[((size_t)(b * 9 + oc) * HH + h) * WW + wpix] =
        2.0f / (1.0f + __expf(-z));
  }
}

// ---------------------------------------------------------------------------
// Kernel 2: deformable-conv GEMM via WMMA.
// One workgroup (128 thr / 4 waves) = 16 consecutive pixels of one image row.
//   Stage 1: 144 bilinear parameter sets -> LDS
//   Stage 2: gather v tile (16 x 576) f16 -> LDS (rows 16B aligned)
//   Stage 3: wave w owns N slice [16w,16w+16); 18x v_wmma_f32_16x16x32_f16
//   Epilogue: +bias, store y, atomic BN partial sums.
// ---------------------------------------------------------------------------
__global__ __launch_bounds__(128) void dcn_deform_wmma_kernel(
    const float* __restrict__ x, const float* __restrict__ off,
    const float* __restrict__ maskb, const _Float16* __restrict__ wh,
    const float* __restrict__ bias, float* __restrict__ y,
    float* __restrict__ sums) {
  __shared__ __align__(16) _Float16 v_lds[TILE_M * KDIM];   // 18 KB
  __shared__ int   p_y0[TILE_M * KTAP];
  __shared__ int   p_x0[TILE_M * KTAP];
  __shared__ float p_w[TILE_M * KTAP * 4];

  const int tid  = threadIdx.x;
  const int tile = blockIdx.x;                 // 0..NTILE-1
  const int b    = tile / (HH * (WW / TILE_M));
  const int rem  = tile % (HH * (WW / TILE_M));
  const int h    = rem / (WW / TILE_M);
  const int w0   = (rem % (WW / TILE_M)) * TILE_M;

  // ---- stage 1: bilinear params (144 of them) ----
  for (int idx = tid; idx < TILE_M * KTAP; idx += 128) {
    int m = idx / KTAP, tap = idx % KTAP;
    int ky = tap / 3, kx = tap % 3;
    int pw = w0 + m;
    const float* offp = off + ((size_t)(b * 18 + 2 * tap) * HH + h) * WW + pw;
    float dy = offp[0];
    float dx = offp[HH * WW];
    float mk = maskb[((size_t)(b * 9 + tap) * HH + h) * WW + pw];
    float py = dy + (float)(ky - 1) + (float)h;
    float px = dx + (float)(kx - 1) + (float)pw;
    float fy = floorf(py), fx = floorf(px);
    float ly = py - fy, lx = px - fx;
    int y0i = (int)fy, x0i = (int)fx;
    float vy0 = (y0i >= 0 && y0i < HH) ? 1.0f : 0.0f;
    float vy1 = (y0i + 1 >= 0 && y0i + 1 < HH) ? 1.0f : 0.0f;
    float vx0 = (x0i >= 0 && x0i < WW) ? 1.0f : 0.0f;
    float vx1 = (x0i + 1 >= 0 && x0i + 1 < WW) ? 1.0f : 0.0f;
    p_y0[idx] = y0i;
    p_x0[idx] = x0i;
    p_w[idx * 4 + 0] = mk * (1.0f - ly) * (1.0f - lx) * vy0 * vx0;
    p_w[idx * 4 + 1] = mk * (1.0f - ly) * lx          * vy0 * vx1;
    p_w[idx * 4 + 2] = mk * ly          * (1.0f - lx) * vy1 * vx0;
    p_w[idx * 4 + 3] = mk * ly          * lx          * vy1 * vx1;
  }
  __syncthreads();

  // ---- stage 2: gather v tile into LDS (f16) ----
  for (int i = tid; i < TILE_M * KDIM; i += 128) {
    int c = i / (TILE_M * KTAP);
    int r = i % (TILE_M * KTAP);
    int m = r / KTAP, tap = r % KTAP;
    int pidx = m * KTAP + tap;
    int y0i = p_y0[pidx], x0i = p_x0[pidx];
    int y0c = min(max(y0i, 0), HH - 1);
    int y1c = min(max(y0i + 1, 0), HH - 1);
    int x0c = min(max(x0i, 0), WW - 1);
    int x1c = min(max(x0i + 1, 0), WW - 1);
    const float* xc = x + ((size_t)(b * CIN + c) * HH) * WW;
    float v00 = xc[y0c * WW + x0c];
    float v01 = xc[y0c * WW + x1c];
    float v10 = xc[y1c * WW + x0c];
    float v11 = xc[y1c * WW + x1c];
    float val = v00 * p_w[pidx * 4 + 0] + v01 * p_w[pidx * 4 + 1] +
                v10 * p_w[pidx * 4 + 2] + v11 * p_w[pidx * 4 + 3];
    v_lds[m * KDIM + c * KTAP + tap] = (_Float16)val;
  }
  __syncthreads();

  // ---- stage 3: WMMA K-loop ----
  const int lane  = tid & 31;
  const int wave  = tid >> 5;
  const int nbase = wave * 16;
  const int col   = lane & 15;
  const int hi8   = (lane >> 4) & 1;          // lanes 16-31 hold K+8
  const _Float16* arow = v_lds + col * KDIM + hi8 * 8;
  const _Float16* brow = wh + (size_t)(nbase + col) * KDIM + hi8 * 8;

  v8f acc = {};
#pragma unroll
  for (int ks = 0; ks < KDIM; ks += 32) {
    union { v16h v; v8h h[2]; } ua, ub;
    ua.h[0] = *(const v8h*)(arow + ks);        // K ks..ks+7   (+8 for hi lanes)
    ua.h[1] = *(const v8h*)(arow + ks + 16);   // K ks+16..23  (+8 for hi lanes)
    ub.h[0] = *(const v8h*)(brow + ks);
    ub.h[1] = *(const v8h*)(brow + ks + 16);
    acc = __builtin_amdgcn_wmma_f32_16x16x32_f16(
        /*neg_a=*/false, ua.v, /*neg_b=*/false, ub.v,
        /*c_mod=*/(short)0, acc, /*reuse_a=*/false, /*reuse_b=*/false);
  }

  // ---- epilogue: bias, store y, BN partial sums ----
  const int o  = nbase + col;                  // output channel (C-layout N=lane%16)
  const int mb = hi8 * 8;                      // C-layout: M = j (+8 for hi lanes)
  float bo = bias[o];
  float s = 0.0f, s2 = 0.0f;
  float* yb = y + ((size_t)(b * COUT + o) * HH + h) * WW + w0;
#pragma unroll
  for (int j = 0; j < 8; ++j) {
    float v = acc[j] + bo;
    yb[mb + j] = v;
    s += v;
    s2 += v * v;
  }
  atomicAdd(&sums[o], s);
  atomicAdd(&sums[COUT + o], s2);
}

// ---------------------------------------------------------------------------
// Kernel 3: BatchNorm (from atomic sums) + affine + ReLU
// ---------------------------------------------------------------------------
__global__ __launch_bounds__(256) void dcn_bnrelu_kernel(
    const float* __restrict__ y, const float* __restrict__ sums,
    const float* __restrict__ gamma, const float* __restrict__ beta,
    float* __restrict__ out) {
  size_t i = (size_t)blockIdx.x * blockDim.x + threadIdx.x;
  const size_t total = (size_t)BATCH * COUT * HH * WW;
  if (i >= total) return;
  int o = (int)((i / (HH * WW)) % COUT);
  const float Nn = (float)(BATCH * HH * WW);
  float mean = sums[o] / Nn;
  float var  = sums[COUT + o] / Nn - mean * mean;
  float inv  = rsqrtf(var + 1e-5f);
  float v = gamma[o] * (y[i] - mean) * inv + beta[o];
  out[i] = v > 0.0f ? v : 0.0f;
}

// ---------------------------------------------------------------------------
extern "C" void kernel_launch(void* const* d_in, const int* in_sizes, int n_in,
                              void* d_out, int out_size, void* d_ws, size_t ws_size,
                              hipStream_t stream) {
  const float* x     = (const float*)d_in[0];
  const float* off_w = (const float*)d_in[1];
  const float* off_b = (const float*)d_in[2];
  const float* mod_w = (const float*)d_in[3];
  const float* mod_b = (const float*)d_in[4];
  const float* w     = (const float*)d_in[5];
  const float* bias  = (const float*)d_in[6];
  const float* gamma = (const float*)d_in[7];
  const float* beta  = (const float*)d_in[8];
  float* out = (float*)d_out;

  // workspace layout (all 256B-ish aligned)
  char* ws = (char*)d_ws;
  _Float16* wh   = (_Float16*)ws;                       // 73728 B
  float*    sums = (float*)(ws + 73728);                //   512 B
  float*    offo = (float*)(ws + 74240);                // 9.0 MB (B*18*H*W)
  float*    mask = (float*)(ws + 74240 + 9437184);      // 4.5 MB (B*9*H*W)
  float*    yb   = (float*)(ws + 74240 + 9437184 + 4718592); // 32 MB
  size_t need = 74240 + 9437184 + 4718592 + (size_t)BATCH * COUT * HH * WW * 4;
  if (ws_size < need) return;

  dcn_prep_kernel<<<(COUT * KDIM + 255) / 256, 256, 0, stream>>>(w, wh, sums);
  dcn_offmask_kernel<<<(NPIX + 255) / 256, 256, 0, stream>>>(
      x, off_w, off_b, mod_w, mod_b, offo, mask);
  dcn_deform_wmma_kernel<<<NTILE, 128, 0, stream>>>(
      x, offo, mask, wh, bias, yb, sums);
  const size_t total = (size_t)BATCH * COUT * HH * WW;
  dcn_bnrelu_kernel<<<(unsigned)((total + 255) / 256), 256, 0, stream>>>(
      yb, sums, gamma, beta, out);
}
